// CommGroup_48515950576058
// MI455X (gfx1250) — matrix operand order repaired and checked by reading, not compiled, not run on hardware
//
#include <hip/hip_runtime.h>
#include <hip/hip_bf16.h>

// ---------------------------------------------------------------------------
// CDNA5 (gfx1250) fused GCM kernel: encoder -> QKV -> masked softmax attention
// -> fused LSTM-gate WMMA GEMM -> decoder. All GEMMs via v_wmma_f32_16x16x32_bf16.
// 32 rows (4 agent groups) per 8-wave block; B fragments reused across 2 row
// tiles in registers to halve L2 weight traffic.
// ---------------------------------------------------------------------------

typedef __bf16 bf16_t;
typedef bf16_t v16bf __attribute__((ext_vector_type(16)));
typedef bf16_t v8bf  __attribute__((ext_vector_type(8)));
typedef float  v8f   __attribute__((ext_vector_type(8)));

#define N_IN   128
#define N_H    256
#define N_A    64
#define N_NA   32
#define N_AG   8        // agents per group
#define ROWS   32       // rows (agents) per block = 4 groups, 2 row tiles
#define KFIN   448      // 128 (mo) + 64 (e) + 256 (hidden)
#define QKV_LD 200      // padded q|k|v row stride (bf16 elements)

// ---- workspace layout (bytes) ----
#define OFF_WENC   0                              // [256][128] bf16
#define OFF_WQKV   (OFF_WENC + 256*128*2)         // [192][256] bf16  (q|k|v rows)
#define OFF_WG     (OFF_WQKV + 192*256*2)         // [1024][448] bf16 ([W_ih|W_hh])
#define OFF_WDEC   (OFF_WG   + 1024*448*2)        // [32][256] bf16
#define OFF_BG     (OFF_WDEC + 32*256*2)          // [1024] f32 (b_ih + b_hh)

static __device__ __forceinline__ float sigf(float x) {
    return 1.f / (1.f + __expf(-x));
}
static __device__ __forceinline__ float tanh_fast(float x) {
    x = fminf(fmaxf(x, -15.f), 15.f);
    float e = __expf(2.f * x);
    return (e - 1.f) / (e + 1.f);
}

static __device__ __forceinline__ v8f wmma_bf16(v16bf a, v16bf b, v8f c) {
    // (neg_a, A, neg_b, B, c_mod, C, reuse_a, reuse_b)
    return __builtin_amdgcn_wmma_f32_16x16x32_bf16(false, a, false, b,
                                                   (short)0, c, false, false);
}

// A fragment: 16x32 bf16, row-major source with leading dim `ld` (elements).
// Contiguous k order (matches load_frag_B): lanes 0-15 row M=lane, K in 0..15; lanes 16-31 same M,
// K in 16..31  ->  two 16B chunks per lane.
static __device__ __forceinline__ v16bf load_frag_A(const bf16_t* base, int ld, int k0) {
    int lane = threadIdx.x & 31;
    int m  = lane & 15;
    int kh = (lane >> 4) << 4;              // 0 or 16
    const bf16_t* p = base + m * ld + k0 + kh;
    v8bf lo = *(const v8bf*)(p);            // K = k0+kh .. +7
    v8bf hi = *(const v8bf*)(p + 8);        // K = k0+kh+8 .. +15
    v16bf r;
#pragma unroll
    for (int i = 0; i < 8; ++i) { r[i] = lo[i]; r[i + 8] = hi[i]; }
    return r;
}

// B fragment: 32x16 bf16 where B[k][n] = W[n0+n][k] and W is row-major [out][in].
// lane holds column n = lane&15, K range 16*(lane>>4)..+15 -> one 32B load.
static __device__ __forceinline__ v16bf load_frag_B(const bf16_t* W, int ld, int n0, int k0) {
    int lane = threadIdx.x & 31;
    const bf16_t* p = W + (n0 + (lane & 15)) * ld + k0 + ((lane >> 4) << 4);
    return *(const v16bf*)p;
}

// ---------------------------------------------------------------------------
// Weight prep: fp32 -> bf16, pack QKV and [W_ih|W_hh], fold biases.
// ---------------------------------------------------------------------------
__global__ void gcm_prep_kernel(const float* __restrict__ W_enc,
                                const float* __restrict__ W_q,
                                const float* __restrict__ W_k,
                                const float* __restrict__ W_v,
                                const float* __restrict__ W_ih,
                                const float* __restrict__ W_hh,
                                const float* __restrict__ b_ih,
                                const float* __restrict__ b_hh,
                                const float* __restrict__ W_dec,
                                bf16_t* __restrict__ wEnc,
                                bf16_t* __restrict__ wQKV,
                                bf16_t* __restrict__ wG,
                                bf16_t* __restrict__ wDec,
                                float*  __restrict__ bg) {
    int idx = blockIdx.x * blockDim.x + threadIdx.x;
    int stride = gridDim.x * blockDim.x;
    for (int i = idx; i < 256 * 128; i += stride)
        wEnc[i] = (bf16_t)W_enc[i];
    for (int i = idx; i < 192 * 256; i += stride) {
        int r = i >> 8, c = i & 255;
        float v = (r < 64) ? W_q[r * 256 + c]
                : (r < 128) ? W_k[(r - 64) * 256 + c]
                            : W_v[(r - 128) * 256 + c];
        wQKV[i] = (bf16_t)v;
    }
    for (int i = idx; i < 1024 * 448; i += stride) {
        int r = i / 448, c = i % 448;
        float v = (c < 192) ? W_ih[r * 192 + c] : W_hh[r * 256 + (c - 192)];
        wG[i] = (bf16_t)v;
    }
    for (int i = idx; i < 32 * 256; i += stride)
        wDec[i] = (bf16_t)W_dec[i];
    for (int i = idx; i < 1024; i += stride)
        bg[i] = b_ih[i] + b_hh[i];
}

// ---------------------------------------------------------------------------
// Fused GCM forward. 256 threads (8 wave32), 32 rows (4 agent groups) / block.
// ---------------------------------------------------------------------------
__global__ __launch_bounds__(256)
void gcm_fused_kernel(const float* __restrict__ d_mo,
                      const float* __restrict__ d_hid,
                      const float* __restrict__ d_cell,
                      const float* __restrict__ b_enc,
                      const float* __restrict__ b_v,
                      const float* __restrict__ b_dec,
                      const bf16_t* __restrict__ wEnc,
                      const bf16_t* __restrict__ wQKV,
                      const bf16_t* __restrict__ wG,
                      const bf16_t* __restrict__ wDec,
                      const float* __restrict__ bg,
                      float* __restrict__ d_m,     // [NB, 32]
                      float* __restrict__ d_h,     // [NB, 256]
                      float* __restrict__ d_c) {   // [NB, 256]
    // sFin: [mo | e | hidden] bf16, lives the whole kernel        (28 KB)
    // sX:   encoder output bf16, stages B->C                      (16 KB)
    // sPool: q|k|v bf16 (stride 200, stages C->D) then h_out bf16
    //        (stride 256, stages E->F); lifetimes disjoint        (16 KB)
    __shared__ __attribute__((aligned(32))) bf16_t sFin[ROWS][KFIN];
    __shared__ __attribute__((aligned(32))) bf16_t sX[ROWS][N_H];
    __shared__ __attribute__((aligned(32))) bf16_t sPool[ROWS * N_H];
    __shared__ float sS[4][N_AG][N_AG];            // attn scores/weights

    const int tid  = threadIdx.x;
    const int wave = tid >> 5;
    const int lane = tid & 31;
    const int n    = lane & 15;     // D-matrix column within tile
    const int hi   = lane >> 4;     // row-half select
    const int row0 = blockIdx.x * ROWS;

    // ---- Prefetch cell_state rows (consumed late, in stage E epilogue) ---
    if (tid < 128) {
        int r = tid >> 2, seg = tid & 3;           // 32 rows x 4 x 256B lines
        __builtin_prefetch(d_cell + (row0 + r) * N_H + seg * 64, 0, 1);
    }

    // ---- Stage A: stage master_out + hidden into sFin (bf16) -------------
    for (int i = tid; i < ROWS * N_IN; i += 256) {
        int r = i >> 7, c = i & 127;
        sFin[r][c] = (bf16_t)d_mo[(row0 + r) * N_IN + c];
    }
    for (int i = tid; i < ROWS * N_H; i += 256) {
        int r = i >> 8, c = i & 255;
        sFin[r][192 + c] = (bf16_t)d_hid[(row0 + r) * N_H + c];
    }
    __syncthreads();

    // ---- Stage B: x = relu(mo @ W_enc^T + b_enc)  [32 x 256] -------------
    {
        v8f acc[2][2];                             // [row tile][n tile]
#pragma unroll
        for (int rt = 0; rt < 2; ++rt)
#pragma unroll
            for (int t = 0; t < 2; ++t)
                acc[rt][t] = (v8f){0.f, 0.f, 0.f, 0.f, 0.f, 0.f, 0.f, 0.f};
#pragma unroll
        for (int k = 0; k < N_IN; k += 32) {
            v16bf a0 = load_frag_A(&sFin[0][0],  KFIN, k);
            v16bf a1 = load_frag_A(&sFin[16][0], KFIN, k);
#pragma unroll
            for (int t = 0; t < 2; ++t) {
                v16bf b = load_frag_B(wEnc, N_IN, (wave * 2 + t) * 16, k);
                acc[0][t] = wmma_bf16(a0, b, acc[0][t]);
                acc[1][t] = wmma_bf16(a1, b, acc[1][t]);
            }
        }
#pragma unroll
        for (int t = 0; t < 2; ++t) {
            int col = (wave * 2 + t) * 16 + n;
            float bb = b_enc[col];
#pragma unroll
            for (int rt = 0; rt < 2; ++rt)
#pragma unroll
                for (int r = 0; r < 8; ++r) {
                    float v = acc[rt][t][r] + bb;
                    sX[rt * 16 + r + 8 * hi][col] = (bf16_t)(v > 0.f ? v : 0.f);
                }
        }
    }
    __syncthreads();

    // ---- Stage C: packed q|k|v = x @ Wqkv^T  [32 x 192] (bf16 out) -------
    for (int idx = wave; idx < 24; idx += 8) {     // wave-uniform loop
        int rt = idx / 12, nt = idx % 12;
        v8f acc = {0.f, 0.f, 0.f, 0.f, 0.f, 0.f, 0.f, 0.f};
#pragma unroll
        for (int k = 0; k < N_H; k += 32)
            acc = wmma_bf16(load_frag_A(&sX[rt * 16][0], N_H, k),
                            load_frag_B(wQKV, N_H, nt * 16, k), acc);
        int col = nt * 16 + n;                     // 0..63 q, 64..127 k, 128..191 v
        float bb = (col >= 128) ? b_v[col - 128] : 0.f;
#pragma unroll
        for (int r = 0; r < 8; ++r) {
            float v = acc[r] + bb;
            if (col >= 128) v = v > 0.f ? v : 0.f;
            sPool[(rt * 16 + r + 8 * hi) * QKV_LD + col] = (bf16_t)v;
        }
    }
    __syncthreads();

    // ---- Stage D: masked softmax attention (4 groups of 8, VALU) ---------
    {                                              // one thread per (g,i,j)
        int g = tid >> 6, i = (tid >> 3) & 7, j = tid & 7;
        const bf16_t* qi = &sPool[(g * 8 + i) * QKV_LD];
        const bf16_t* kj = &sPool[(g * 8 + j) * QKV_LD + 64];
        float dot = 0.f;
#pragma unroll
        for (int a = 0; a < N_A; ++a)
            dot += (float)qi[a] * (float)kj[a];
        sS[g][i][j] = (i == j) ? -1e9f : dot * 0.125f;   // 1/sqrt(64)
    }
    __syncthreads();
    if (tid < 32) {                                // one thread per (g,i): softmax
        int g = tid >> 3, i = tid & 7;
        float mx = -1e30f;
#pragma unroll
        for (int j = 0; j < 8; ++j) mx = fmaxf(mx, sS[g][i][j]);
        float sum = 0.f, e[8];
#pragma unroll
        for (int j = 0; j < 8; ++j) { e[j] = __expf(sS[g][i][j] - mx); sum += e[j]; }
        float inv = 1.f / sum;
#pragma unroll
        for (int j = 0; j < 8; ++j) sS[g][i][j] = e[j] * inv;
    }
    __syncthreads();
#pragma unroll
    for (int it = 0; it < 8; ++it) {               // e = w @ v -> sFin[:,128:192]
        int eid = tid + 256 * it;                  // 32*64 elements
        int r = eid >> 6, a = eid & 63, g = r >> 3, ii = r & 7;
        float sum = 0.f;
#pragma unroll
        for (int j = 0; j < 8; ++j)
            sum += sS[g][ii][j] * (float)sPool[(g * 8 + j) * QKV_LD + 128 + a];
        sFin[r][128 + a] = (bf16_t)sum;
    }
    __syncthreads();                               // also retires sPool=qkv use

    // ---- Stage E: fused LSTM gates: [mo|e|h] @ [W_ih|W_hh]^T, K=448 ------
    // Wave owns hidden tiles {wave, wave+8}; 4 gate accumulators per tile so
    // i/f/g/o for one hidden unit live in the same lane; each B fragment is
    // reused for both row tiles (halves L2 weight traffic).
    {
        v8f acc[2][2][4];                          // [row tile][ht][gate]
#pragma unroll
        for (int rt = 0; rt < 2; ++rt)
#pragma unroll
            for (int t = 0; t < 2; ++t)
#pragma unroll
                for (int g = 0; g < 4; ++g)
                    acc[rt][t][g] = (v8f){0.f, 0.f, 0.f, 0.f, 0.f, 0.f, 0.f, 0.f};

        for (int k = 0; k < KFIN; k += 32) {
            v16bf a0 = load_frag_A(&sFin[0][0],  KFIN, k);
            v16bf a1 = load_frag_A(&sFin[16][0], KFIN, k);
#pragma unroll
            for (int t = 0; t < 2; ++t) {
                int ht = wave + 8 * t;
#pragma unroll
                for (int g = 0; g < 4; ++g) {
                    v16bf b = load_frag_B(wG, KFIN, g * N_H + ht * 16, k);
                    acc[0][t][g] = wmma_bf16(a0, b, acc[0][t][g]);
                    acc[1][t][g] = wmma_bf16(a1, b, acc[1][t][g]);
                }
            }
        }
#pragma unroll
        for (int t = 0; t < 2; ++t) {
            int j = (wave + 8 * t) * 16 + n;       // hidden unit index
            float bi = bg[j], bf_ = bg[256 + j], bgg = bg[512 + j], bo = bg[768 + j];
#pragma unroll
            for (int rt = 0; rt < 2; ++rt)
#pragma unroll
                for (int r = 0; r < 8; ++r) {
                    int m = rt * 16 + r + 8 * hi;
                    int grow = row0 + m;
                    float gi = acc[rt][t][0][r] + bi;
                    float gf = acc[rt][t][1][r] + bf_;
                    float gg = acc[rt][t][2][r] + bgg;
                    float go = acc[rt][t][3][r] + bo;
                    float cp = d_cell[grow * N_H + j];
                    float cn = sigf(gf) * cp + sigf(gi) * tanh_fast(gg);
                    float hn = sigf(go) * tanh_fast(cn);
                    d_c[grow * N_H + j] = cn;
                    d_h[grow * N_H + j] = hn;
                    sPool[m * N_H + j] = (bf16_t)hn;   // h_out view, stride 256
                }
        }
    }
    __syncthreads();

    // ---- Stage F: m_actions = h_new @ W_dec^T + b_dec  [32 x 32] ---------
    if (wave < 4) {
        int nt = wave & 1, rt = wave >> 1;
        v8f acc2 = {0.f, 0.f, 0.f, 0.f, 0.f, 0.f, 0.f, 0.f};
#pragma unroll
        for (int k = 0; k < N_H; k += 32)
            acc2 = wmma_bf16(load_frag_A(&sPool[rt * 16 * N_H], N_H, k),
                             load_frag_B(wDec, N_H, nt * 16, k), acc2);
        int col = nt * 16 + n;
        float bb = b_dec[col];
#pragma unroll
        for (int r = 0; r < 8; ++r) {
            int m = rt * 16 + r + 8 * hi;
            d_m[(row0 + m) * N_NA + col] = acc2[r] + bb;
        }
    }
}

extern "C" void kernel_launch(void* const* d_in, const int* in_sizes, int n_in,
                              void* d_out, int out_size, void* d_ws, size_t ws_size,
                              hipStream_t stream) {
    const float* master_out   = (const float*)d_in[0];
    const float* hidden_state = (const float*)d_in[1];
    const float* cell_state   = (const float*)d_in[2];
    const float* W_enc        = (const float*)d_in[3];
    const float* b_enc        = (const float*)d_in[4];
    const float* W_q          = (const float*)d_in[5];
    const float* W_k          = (const float*)d_in[6];
    const float* W_v          = (const float*)d_in[7];
    const float* b_v          = (const float*)d_in[8];
    const float* W_ih         = (const float*)d_in[9];
    const float* W_hh         = (const float*)d_in[10];
    const float* b_ih         = (const float*)d_in[11];
    const float* b_hh         = (const float*)d_in[12];
    const float* W_dec        = (const float*)d_in[13];
    const float* b_dec        = (const float*)d_in[14];

    char* ws = (char*)d_ws;
    bf16_t* wEnc  = (bf16_t*)(ws + OFF_WENC);
    bf16_t* wQKV  = (bf16_t*)(ws + OFF_WQKV);
    bf16_t* wG    = (bf16_t*)(ws + OFF_WG);
    bf16_t* wDec  = (bf16_t*)(ws + OFF_WDEC);
    float*  bgbuf = (float*)(ws + OFF_BG);

    const int NB = in_sizes[0] / N_IN;       // 131072
    float* d_m = (float*)d_out;              // [NB, 32]
    float* d_h = d_m + (size_t)NB * N_NA;    // [NB, 256]
    float* d_c = d_h + (size_t)NB * N_H;     // [NB, 256]

    gcm_prep_kernel<<<512, 256, 0, stream>>>(W_enc, W_q, W_k, W_v, W_ih, W_hh,
                                             b_ih, b_hh, W_dec,
                                             wEnc, wQKV, wG, wDec, bgbuf);

    gcm_fused_kernel<<<NB / ROWS, 256, 0, stream>>>(
        master_out, hidden_state, cell_state, b_enc, b_v, b_dec,
        wEnc, wQKV, wG, wDec, bgbuf, d_m, d_h, d_c);
}
